// GINEModel_43946105373264
// MI455X (gfx1250) — compile-verified, matrix operand order checked by reading, not run on hardware
//
#include <hip/hip_runtime.h>

typedef __attribute__((ext_vector_type(16))) _Float16 v16h;
typedef __attribute__((ext_vector_type(8)))  float    v8f;
typedef __attribute__((ext_vector_type(4)))  float    f4;

__device__ __forceinline__ v8f wmma16(v16h a, v16h b, v8f c) {
  // D = A(16x32 f16) * B(32x16 f16) + C(16x16 f32)
  return __builtin_amdgcn_wmma_f32_16x16x32_f16(false, a, false, b, (short)0, c,
                                                false, false);
}

// ---------------------------------------------------------------------------
// h = x @ node_W + node_b   ([N,32] x [32,64]); one WMMA per 16x16 tile (K=32)
// Weights transposed in LDS (N-major) -> contiguous b128 DS loads for B.
// Output staged in LDS -> coalesced b128 global stores, one guard per lane.
// ---------------------------------------------------------------------------
__global__ void __launch_bounds__(256) node_enc_kernel(
    const float* __restrict__ x, const float* __restrict__ W,
    const float* __restrict__ bias, float* __restrict__ h, int n)
{
  __shared__ _Float16 sWt[64 * 32];       // [n][k]
  __shared__ float    sO[8][16 * 64];     // per-wave f32 output staging
  int tid = threadIdx.x;
#pragma unroll
  for (int i = 0; i < 8; i++) {
    int j = tid + i * 256;                // j = k*64 + n
    sWt[(j & 63) * 32 + (j >> 6)] = (_Float16)W[j];
  }
  __syncthreads();

  int wave = tid >> 5, lane = tid & 31;
  int r16 = lane & 15, kh = (lane >> 4) * 8;
  int rowbase = blockIdx.x * 128 + wave * 16;
  int arow = rowbase + r16; if (arow >= n) arow = n - 1;

  // A fragment: K runs [kh, kh+8) and [16+kh, 16+kh+8) -> four b128 loads
  const f4* xp4 = (const f4*)(x + (size_t)arow * 32);
  int b4 = kh >> 2;                       // 0 or 2
  f4 xc[4];
  xc[0] = xp4[b4];     xc[1] = xp4[b4 + 1];
  xc[2] = xp4[b4 + 4]; xc[3] = xp4[b4 + 5];
  v16h a;
#pragma unroll
  for (int i = 0; i < 8; i++) {
    a[i]     = (_Float16)xc[i >> 2][i & 3];
    a[i + 8] = (_Float16)xc[2 + (i >> 2)][i & 3];
  }

#pragma unroll
  for (int t = 0; t < 4; t++) {
    int ncol = t * 16 + r16;
    const _Float16* wc = &sWt[ncol * 32];
    v16h bb;
#pragma unroll
    for (int i = 0; i < 8; i++) {
      bb[i]     = wc[kh + i];             // contiguous -> ds_load_b128
      bb[i + 8] = wc[16 + kh + i];
    }
    v8f acc = {};
    acc = wmma16(a, bb, acc);
    float bv = bias[ncol];
#pragma unroll
    for (int r = 0; r < 8; r++) sO[wave][(kh + r) * 64 + ncol] = acc[r] + bv;
  }
  __syncthreads();

  // coalesced store: each lane owns half a row (32 contiguous floats)
  int orow = lane >> 1, ocol = (lane & 1) * 32;
  int grow = rowbase + orow;
  if (grow < n) {
    f4* dst = (f4*)(h + (size_t)grow * 64 + ocol);
    const f4* src = (const f4*)&sO[wave][orow * 64 + ocol];
#pragma unroll
    for (int q = 0; q < 8; q++) dst[q] = src[q];
  }
}

// ---------------------------------------------------------------------------
// Fused: e_ij = edge_attr @ edge_W + edge_b (K=16, trivial FLOPs, recomputed
// per layer so the 410MB e-buffer never exists) ; msg = relu(h[src]+e_ij) ;
// atomic scatter into agg[dst]. h/agg are L2-resident (<192MB global L2).
// ---------------------------------------------------------------------------
__global__ void __launch_bounds__(256) scatter_kernel(
    const float* __restrict__ h, const float* __restrict__ edge_attr,
    const int* __restrict__ ei, const float* __restrict__ eW,
    const float* __restrict__ eB, float* __restrict__ agg, int n_edges)
{
  __shared__ float sWt[64 * 16];          // [c][k] transposed
  __shared__ float sB[64];
  int tid = threadIdx.x;
#pragma unroll
  for (int i = 0; i < 4; i++) {
    int j = tid + i * 256;                // j = k*64 + c
    sWt[(j & 63) * 16 + (j >> 6)] = eW[j];
  }
  if (tid < 64) sB[tid] = eB[tid];
  __syncthreads();

  int stride = gridDim.x * blockDim.x;
  for (int edge = blockIdx.x * blockDim.x + tid; edge < n_edges; edge += stride) {
    const f4* ap = (const f4*)(edge_attr + (size_t)edge * 16);
    f4 ea[4];
#pragma unroll
    for (int q = 0; q < 4; q++) ea[q] = ap[q];
    int s = ei[edge];
    int d = ei[n_edges + edge];
    const f4* hs = (const f4*)(h + (size_t)s * 64);
    float* ad = agg + (size_t)d * 64;
#pragma unroll 2
    for (int c4 = 0; c4 < 16; c4++) {
      f4 hv = hs[c4];
#pragma unroll
      for (int u = 0; u < 4; u++) {
        int c = c4 * 4 + u;
        const float* wr = &sWt[c * 16];
        float v = sB[c];
#pragma unroll
        for (int k = 0; k < 16; k++) v += ea[k >> 2][k & 3] * wr[k];
        v += hv[u];
        v = fmaxf(v, 0.0f);
        atomicAdd(&ad[c], v);
      }
    }
  }
}

// ---------------------------------------------------------------------------
// z = relu((h+agg) @ W1 + b1) @ W2 + b2 ; both 64x64 GEMMs via WMMA.
// Transposed LDS weights (b128 B loads), LDS relayout between GEMMs,
// LDS-staged f32 output with coalesced b128 stores.
// ---------------------------------------------------------------------------
__global__ void __launch_bounds__(256) mlp_kernel(
    const float* __restrict__ h, const float* __restrict__ agg,
    const float* __restrict__ W1, const float* __restrict__ bias1,
    const float* __restrict__ W2, const float* __restrict__ bias2,
    float* __restrict__ z, int n)
{
  __shared__ _Float16 sW1t[64 * 64];      // [n][k]
  __shared__ _Float16 sW2t[64 * 64];      // [n][k]
  __shared__ _Float16 sZ[8][16 * 64];     // f16 intermediate (A for GEMM2)
  __shared__ float    sO[8][16 * 64];     // f32 output staging
  int tid = threadIdx.x;
#pragma unroll
  for (int i = 0; i < 16; i++) {
    int j = tid + i * 256;                // j = k*64 + n
    int tr = (j & 63) * 64 + (j >> 6);
    sW1t[tr] = (_Float16)W1[j];
    sW2t[tr] = (_Float16)W2[j];
  }
  __syncthreads();

  int wave = tid >> 5, lane = tid & 31;
  int r16 = lane & 15, kh = (lane >> 4) * 8;
  int rowbase = blockIdx.x * 128 + wave * 16;
  int arow = rowbase + r16; if (arow >= n) arow = n - 1;

  // A fragments for GEMM1: eight b128 loads each from h and agg
  const f4* hp4 = (const f4*)(h + (size_t)arow * 64);
  const f4* gp4 = (const f4*)(agg + (size_t)arow * 64);
  int b4 = kh >> 2;                       // 0 or 2
  f4 hc[8], gc[8];
#pragma unroll
  for (int q = 0; q < 4; q++) {
    hc[2 * q]     = hp4[b4 + 4 * q];
    hc[2 * q + 1] = hp4[b4 + 4 * q + 1];
    gc[2 * q]     = gp4[b4 + 4 * q];
    gc[2 * q + 1] = gp4[b4 + 4 * q + 1];
  }
  v16h a0, a1;
#pragma unroll
  for (int i = 0; i < 8; i++) {
    a0[i]     = (_Float16)(hc[i >> 2][i & 3]       + gc[i >> 2][i & 3]);
    a0[i + 8] = (_Float16)(hc[2 + (i >> 2)][i & 3] + gc[2 + (i >> 2)][i & 3]);
    a1[i]     = (_Float16)(hc[4 + (i >> 2)][i & 3] + gc[4 + (i >> 2)][i & 3]);
    a1[i + 8] = (_Float16)(hc[6 + (i >> 2)][i & 3] + gc[6 + (i >> 2)][i & 3]);
  }

  _Float16* zrow = &sZ[wave][0];
#pragma unroll
  for (int t = 0; t < 4; t++) {
    int ncol = t * 16 + r16;
    const _Float16* wc = &sW1t[ncol * 64];
    v16h b0, b1v;
#pragma unroll
    for (int i = 0; i < 8; i++) {
      b0[i]      = wc[kh + i];            // contiguous runs -> ds_load_b128
      b0[i + 8]  = wc[16 + kh + i];
      b1v[i]     = wc[32 + kh + i];
      b1v[i + 8] = wc[48 + kh + i];
    }
    v8f acc = {};
    acc = wmma16(a0, b0, acc);
    acc = wmma16(a1, b1v, acc);
    float bv = bias1[ncol];
#pragma unroll
    for (int r = 0; r < 8; r++) {
      float v = fmaxf(acc[r] + bv, 0.0f);          // relu between GEMMs
      zrow[(kh + r) * 64 + ncol] = (_Float16)v;    // D-layout -> LDS
    }
  }
  __syncthreads();

  // A fragments for GEMM2 from staged intermediate (contiguous 8-half runs)
  v16h c0, c1;
#pragma unroll
  for (int i = 0; i < 8; i++) {
    c0[i]     = zrow[r16 * 64 + kh + i];
    c0[i + 8] = zrow[r16 * 64 + 16 + kh + i];
    c1[i]     = zrow[r16 * 64 + 32 + kh + i];
    c1[i + 8] = zrow[r16 * 64 + 48 + kh + i];
  }
#pragma unroll
  for (int t = 0; t < 4; t++) {
    int ncol = t * 16 + r16;
    const _Float16* wc = &sW2t[ncol * 64];
    v16h b0, b1v;
#pragma unroll
    for (int i = 0; i < 8; i++) {
      b0[i]      = wc[kh + i];
      b0[i + 8]  = wc[16 + kh + i];
      b1v[i]     = wc[32 + kh + i];
      b1v[i + 8] = wc[48 + kh + i];
    }
    v8f acc = {};
    acc = wmma16(c0, b0, acc);
    acc = wmma16(c1, b1v, acc);
    float bv = bias2[ncol];
#pragma unroll
    for (int r = 0; r < 8; r++) sO[wave][(kh + r) * 64 + ncol] = acc[r] + bv;
  }
  __syncthreads();

  // coalesced store of z: each lane owns 32 contiguous floats of one row
  int orow = lane >> 1, ocol = (lane & 1) * 32;
  int grow = rowbase + orow;
  if (grow < n) {
    f4* dst = (f4*)(z + (size_t)grow * 64 + ocol);
    const f4* src = (const f4*)&sO[wave][orow * 64 + ocol];
#pragma unroll
    for (int q = 0; q < 8; q++) dst[q] = src[q];
  }
}

// ---------------------------------------------------------------------------
// BatchNorm statistics: each thread owns a fixed 4-channel group (stride in
// f4 units is a multiple of 16), b128 streaming loads, 8 atomics per thread.
// ---------------------------------------------------------------------------
__global__ void __launch_bounds__(256) bn_stats_kernel(
    const float* __restrict__ z, float* __restrict__ bn, int n)
{
  size_t total  = (size_t)n * 16;                    // in f4 units
  size_t stride = (size_t)gridDim.x * blockDim.x;    // multiple of 16
  size_t i0 = (size_t)blockIdx.x * blockDim.x + threadIdx.x;
  int cg = (int)(i0 & 15);
  const f4* z4 = (const f4*)z;
  f4 s = {0.f, 0.f, 0.f, 0.f}, s2 = {0.f, 0.f, 0.f, 0.f};
  for (size_t i = i0; i < total; i += stride) {
    f4 v = z4[i];
    s += v; s2 += v * v;
  }
#pragma unroll
  for (int u = 0; u < 4; u++) {
    atomicAdd(&bn[cg * 4 + u], s[u]);
    atomicAdd(&bn[64 + cg * 4 + u], s2[u]);
  }
}

__global__ void __launch_bounds__(256) bn_apply_kernel(
    const float* __restrict__ z, const float* __restrict__ bn,
    const float* __restrict__ gamma, const float* __restrict__ beta,
    float* __restrict__ h, int n)
{
  size_t total  = (size_t)n * 16;
  size_t stride = (size_t)gridDim.x * blockDim.x;
  float inv_n = 1.0f / (float)n;
  const f4* z4 = (const f4*)z;
  f4* h4 = (f4*)h;
  for (size_t i = (size_t)blockIdx.x * blockDim.x + threadIdx.x; i < total; i += stride) {
    int cg = (int)(i & 15);
    f4 v = z4[i], o;
#pragma unroll
    for (int u = 0; u < 4; u++) {
      int c = cg * 4 + u;
      float mu  = bn[c] * inv_n;
      float var = bn[64 + c] * inv_n - mu * mu;      // biased variance
      o[u] = fmaxf((v[u] - mu) * rsqrtf(var + 1e-5f) * gamma[c] + beta[c], 0.0f);
    }
    h4[i] = o;
  }
}

__global__ void zero_kernel(float* __restrict__ p, size_t n) {  // n % 4 == 0
  size_t total  = n >> 2;
  size_t stride = (size_t)gridDim.x * blockDim.x;
  f4 zv = {0.f, 0.f, 0.f, 0.f};
  f4* p4 = (f4*)p;
  for (size_t i = (size_t)blockIdx.x * blockDim.x + threadIdx.x; i < total; i += stride)
    p4[i] = zv;
}

__global__ void __launch_bounds__(256) pool_kernel(
    const float* __restrict__ h, const int* __restrict__ batch,
    float* __restrict__ pool, float* __restrict__ cnt, int n)
{
  size_t total  = (size_t)n * 16;
  size_t stride = (size_t)gridDim.x * blockDim.x;
  const f4* h4 = (const f4*)h;
  for (size_t i = (size_t)blockIdx.x * blockDim.x + threadIdx.x; i < total; i += stride) {
    int node = (int)(i >> 4);
    int cg   = (int)(i & 15);
    int g    = batch[node];
    f4 v = h4[i];
#pragma unroll
    for (int u = 0; u < 4; u++) atomicAdd(&pool[(size_t)g * 64 + cg * 4 + u], v[u]);
    if (cg == 0) atomicAdd(&cnt[g], 1.0f);
  }
}

// One block per graph: g=[sum|mean] (128) -> 256 -> 128 -> 1  (tiny)
__global__ void __launch_bounds__(256) head_kernel(
    const float* __restrict__ pool, const float* __restrict__ cnt,
    const float* __restrict__ W1, const float* __restrict__ b1,
    const float* __restrict__ W2, const float* __restrict__ b2,
    const float* __restrict__ W3, const float* __restrict__ b3,
    float* __restrict__ out)
{
  __shared__ float sg[128];
  __shared__ float s1[256];
  __shared__ float s2[128];
  __shared__ float red[128];
  int g = blockIdx.x, tid = threadIdx.x;
  if (tid < 64)       sg[tid] = pool[(size_t)g * 64 + tid];
  else if (tid < 128) sg[tid] = pool[(size_t)g * 64 + (tid - 64)] / fmaxf(cnt[g], 1.0f);
  __syncthreads();

  float acc = b1[tid];
  for (int k = 0; k < 128; k++) acc += sg[k] * W1[k * 256 + tid];
  s1[tid] = fmaxf(acc, 0.0f);
  __syncthreads();

  if (tid < 128) {
    float a2 = b2[tid];
    for (int k = 0; k < 256; k++) a2 += s1[k] * W2[k * 128 + tid];
    s2[tid] = fmaxf(a2, 0.0f);
  }
  __syncthreads();

  if (tid < 128) red[tid] = s2[tid] * W3[tid];
  __syncthreads();
  for (int off = 64; off > 0; off >>= 1) {
    if (tid < off) red[tid] += red[tid + off];
    __syncthreads();
  }
  if (tid == 0) out[g] = red[0] + b3[0];
}

extern "C" void kernel_launch(void* const* d_in, const int* in_sizes, int n_in,
                              void* d_out, int out_size, void* d_ws, size_t ws_size,
                              hipStream_t stream)
{
  const float* x         = (const float*)d_in[0];
  const float* edge_attr = (const float*)d_in[1];
  const int*   ei        = (const int*)d_in[2];
  const int*   batch     = (const int*)d_in[3];
  const float* node_W    = (const float*)d_in[4];
  const float* node_b    = (const float*)d_in[5];
  const float* edge_W    = (const float*)d_in[6];
  const float* edge_b    = (const float*)d_in[7];
  const float* mlp1_W    = (const float*)d_in[8];
  const float* mlp1_b    = (const float*)d_in[9];
  const float* mlp2_W    = (const float*)d_in[10];
  const float* mlp2_b    = (const float*)d_in[11];
  const float* bn_gamma  = (const float*)d_in[12];
  const float* bn_beta   = (const float*)d_in[13];
  const float* h1W = (const float*)d_in[14];
  const float* h1b = (const float*)d_in[15];
  const float* h2W = (const float*)d_in[16];
  const float* h2b = (const float*)d_in[17];
  const float* h3W = (const float*)d_in[18];
  const float* h3b = (const float*)d_in[19];
  float* out = (float*)d_out;

  int n_nodes  = in_sizes[3];        // batch vector length
  int n_edges  = in_sizes[2] / 2;    // edge_index is [2, E]
  int n_graphs = out_size;           // 128

  // Workspace: h, agg, z (each N*64 f32, all L2-resident) + bn + pool + cnt
  float* ws = (float*)d_ws;
  size_t off = 0;
  float* h    = ws + off; off += (size_t)n_nodes * 64;
  float* agg  = ws + off; off += (size_t)n_nodes * 64;
  float* z    = ws + off; off += (size_t)n_nodes * 64;
  float* bn   = ws + off; off += 128;
  float* pool = ws + off; off += (size_t)n_graphs * 64;
  float* cnt  = ws + off; off += n_graphs;
  (void)n_in; (void)ws_size;

  int rowBlocks = (n_nodes + 127) / 128;   // 128 rows per block (8 waves x 16)
  node_enc_kernel<<<rowBlocks, 256, 0, stream>>>(x, node_W, node_b, h, n_nodes);

  for (int l = 0; l < 4; l++) {
    zero_kernel<<<1024, 256, 0, stream>>>(agg, (size_t)n_nodes * 64);
    zero_kernel<<<1, 256, 0, stream>>>(bn, (size_t)128);

    int eb = (n_edges + 255) / 256; if (eb > 8192) eb = 8192;
    scatter_kernel<<<eb, 256, 0, stream>>>(h, edge_attr, ei, edge_W, edge_b,
                                           agg, n_edges);

    mlp_kernel<<<rowBlocks, 256, 0, stream>>>(
        h, agg,
        mlp1_W + (size_t)l * 64 * 64, mlp1_b + (size_t)l * 64,
        mlp2_W + (size_t)l * 64 * 64, mlp2_b + (size_t)l * 64,
        z, n_nodes);

    bn_stats_kernel<<<1024, 256, 0, stream>>>(z, bn, n_nodes);
    bn_apply_kernel<<<2048, 256, 0, stream>>>(z, bn, bn_gamma + l * 64,
                                              bn_beta + l * 64, h, n_nodes);
  }

  zero_kernel<<<64, 256, 0, stream>>>(pool, (size_t)(n_graphs * 64 + n_graphs));
  pool_kernel<<<2048, 256, 0, stream>>>(h, batch, pool, cnt, n_nodes);
  head_kernel<<<n_graphs, 256, 0, stream>>>(pool, cnt, h1W, h1b, h2W, h2b,
                                            h3W, h3b, out);
}
